// NonLocalLayer_35845797052457
// MI455X (gfx1250) — matrix operand order repaired and checked by reading, not compiled, not run on hardware
//
#include <hip/hip_runtime.h>
#include <hip/hip_bf16.h>

typedef __attribute__((ext_vector_type(16))) _Float16 v16h;
typedef __attribute__((ext_vector_type(8)))  _Float16 v8h;
typedef __attribute__((ext_vector_type(8)))  float    v8f;

// Problem constants (from reference: B=4, C=256, H=W=64)
constexpr int BB = 4;
constexpr int C  = 256;
constexpr int C2 = 128;
constexpr int NN = 4096;   // H*W

// ---------------------------------------------------------------------------
// CDNA5 async copy: global -> LDS, 16 bytes per lane, tracked by ASYNCcnt.
// lds_off is the addrspace(3) byte offset (low 32 bits of a generic pointer
// to __shared__ memory equal the wave-relative LDS address per the aperture
// truncation rule).
// ---------------------------------------------------------------------------
__device__ __forceinline__ void async_copy_b128(unsigned lds_off, const void* gaddr) {
    asm volatile("global_load_async_to_lds_b128 %0, %1, off"
                 :: "v"(lds_off), "v"(gaddr) : "memory");
}
__device__ __forceinline__ void wait_async0() {
    asm volatile("s_wait_asynccnt 0x0" ::: "memory");
}

// ---------------------------------------------------------------------------
// Kernel 0: convert fp32 weights -> f16 in workspace.
// Layout in ws (halves): [0,32768) w_theta, [32768,65536) w_phi,
// [65536,98304) w_g, [98304,131072) w_out.
// ---------------------------------------------------------------------------
__global__ void cvt_weights_kernel(const float* __restrict__ wt,
                                   const float* __restrict__ wp,
                                   const float* __restrict__ wg,
                                   const float* __restrict__ wo,
                                   _Float16* __restrict__ out) {
    int i = blockIdx.x * blockDim.x + threadIdx.x;
    if (i < 32768)            out[i] = (_Float16)wt[i];
    else if (i < 65536)       out[i] = (_Float16)wp[i - 32768];
    else if (i < 98304)       out[i] = (_Float16)wg[i - 65536];
    else if (i < 131072)      out[i] = (_Float16)wo[i - 98304];
}

// ---------------------------------------------------------------------------
// Kernel 1: projection GEMM.  theta/phi stored [B][N][C2]; g stored [B][C2][N]
// (pre-transposed for the attention kernel's V tiles).
// grid = (N/64, 3, B), block = 128 (4 waves).
// x chunks (32 channels x 64 pixels, f32) double-buffered in LDS via async
// global->LDS copies; f32->f16 conversion happens in the A-fragment gather.
// ---------------------------------------------------------------------------
__global__ __launch_bounds__(128)
void proj_kernel(const float* __restrict__ x,
                 const _Float16* __restrict__ w16,   // [3][C2][C]
                 const float* __restrict__ b_theta,
                 const float* __restrict__ b_phi,
                 const float* __restrict__ b_g,
                 _Float16* __restrict__ qkv) {       // [3][B][...]
    const int n0   = blockIdx.x * 64;
    const int proj = blockIdx.y;
    const int b    = blockIdx.z;

    const _Float16* w = w16 + (size_t)proj * C2 * C;
    const float* bias = (proj == 0) ? b_theta : (proj == 1) ? b_phi : b_g;
    _Float16* out = qkv + ((size_t)proj * BB + b) * NN * C2;

    // [buf][c_local][n_local], row stride 68 floats = 272B (16B multiple)
    __shared__ __align__(16) float xs32[2][32][68];

    const int wave = threadIdx.x >> 5;
    const int lane = threadIdx.x & 31;
    const int g    = lane >> 4;       // lane half-group
    const int l15  = lane & 15;
    const int nw   = n0 + wave * 16;  // this wave's row base (global n)

    const unsigned xs_base = (unsigned)(uintptr_t)&xs32[0][0][0];

    auto stage_x = [&](int c0, int buf) {
        const int cl = threadIdx.x >> 2;          // 0..31
        const int nb = (threadIdx.x & 3) * 16;    // 0,16,32,48
        const float* src = x + ((size_t)b * C + c0 + cl) * NN + n0 + nb;
        const unsigned l = xs_base + (unsigned)(buf * (32 * 68 * 4)
                                               + cl * (68 * 4) + nb * 4);
        #pragma unroll
        for (int i = 0; i < 4; ++i)
            async_copy_b128(l + 16 * i, src + 4 * i);
    };

    v8f acc[8] = {};

    stage_x(0, 0);   // prologue prefetch

    for (int c0 = 0; c0 < C; c0 += 32) {
        const int buf = (c0 >> 5) & 1;
        wait_async0();
        __syncthreads();
        if (c0 + 32 < C) stage_x(c0 + 32, buf ^ 1);

        // A fragment (16x32 f16): lane row = l15; K contiguous per lane half (k = 16*g + s);
        // gather f32 from [c][n] LDS and convert.
        v16h a;
        const int arow = wave * 16 + l15;
        #pragma unroll
        for (int j = 0; j < 4; ++j) {
            int kk = 16 * g + 2 * j;
            a[2 * j]     = (_Float16)xs32[buf][kk][arow];
            a[2 * j + 1] = (_Float16)xs32[buf][kk + 1][arow];
            int kh = 16 * g + 8 + 2 * j;
            a[8 + 2 * j]     = (_Float16)xs32[buf][kh][arow];
            a[8 + 2 * j + 1] = (_Float16)xs32[buf][kh + 1][arow];
        }

        // 8 m-tiles; B fragment = 16 contiguous f16 weights per lane
        #pragma unroll
        for (int t = 0; t < 8; ++t) {
            const _Float16* wp = w + (size_t)(t * 16 + l15) * C + c0 + 16 * g;
            v16h bf;
            #pragma unroll
            for (int j = 0; j < 16; ++j) bf[j] = wp[j];
            acc[t] = __builtin_amdgcn_wmma_f32_16x16x32_f16(
                false, a, false, bf, (short)0, acc[t], false, false);
        }
    }

    if (proj == 2) {
        // V: store transposed [C2][N]; rows r are contiguous in n -> 16B stores
        #pragma unroll
        for (int t = 0; t < 8; ++t) {
            const int m = t * 16 + l15;
            const float bia = bias[m];
            v8h hv;
            #pragma unroll
            for (int r = 0; r < 8; ++r) hv[r] = (_Float16)(acc[t][r] + bia);
            *(v8h*)(out + (size_t)m * NN + nw + 8 * g) = hv;
        }
    } else {
        // Q/K: store [N][C2]; lanes cover contiguous m -> coalesced b16 stores
        #pragma unroll
        for (int t = 0; t < 8; ++t) {
            const int m = t * 16 + l15;
            const float bia = bias[m];
            #pragma unroll
            for (int r = 0; r < 8; ++r) {
                const int n = nw + r + 8 * g;
                out[(size_t)n * C2 + m] = (_Float16)(acc[t][r] + bia);
            }
        }
    }
}

// ---------------------------------------------------------------------------
// Kernel 2: fused flash attention + output projection + residual.
// grid = (N/64, B), block = 128 (4 waves). Wave owns 16 query rows.
// K/V chunks (32 keys) double-buffered in LDS via async global->LDS copies;
// online softmax; O kept in WMMA accumulators; O @ w_out^T fused with bias +
// residual store into y[B,C,N]. Per-wave LDS transposes use wave-local
// ordering (no block barrier needed).
// ---------------------------------------------------------------------------
__global__ __launch_bounds__(128)
void attn_kernel(const float* __restrict__ x,
                 const _Float16* __restrict__ qkv,
                 const _Float16* __restrict__ wo16,  // [C][C2] f16
                 const float* __restrict__ b_out,
                 float* __restrict__ y) {
    const int b  = blockIdx.y;
    const int n0 = blockIdx.x * 64;

    const _Float16* q  = qkv + ((size_t)0 * BB + b) * NN * C2;  // [N][C2]
    const _Float16* k  = qkv + ((size_t)1 * BB + b) * NN * C2;  // [N][C2]
    const _Float16* vT = qkv + ((size_t)2 * BB + b) * NN * C2;  // [C2][N]

    // double-buffered tiles; row strides kept 16B-multiples for async b128
    __shared__ __align__(16) _Float16 ks [2][32][136];    // [buf][key][c]
    __shared__ __align__(16) _Float16 vts[2][C2][32];     // [buf][c][key]
    __shared__ __align__(16) _Float16 ps [4][16][34];     // per-wave P bounce
    __shared__ __align__(16) _Float16 os [4][16][C2 + 4]; // per-wave O bounce

    const int wave = threadIdx.x >> 5;
    const int lane = threadIdx.x & 31;
    const int g    = lane >> 4;
    const int l15  = lane & 15;
    const int nw   = n0 + wave * 16;

    const unsigned ks_base  = (unsigned)(uintptr_t)&ks[0][0][0];
    const unsigned vts_base = (unsigned)(uintptr_t)&vts[0][0][0];

    // issue async staging of one 32-key chunk into buffer `buf`
    auto issue_chunk = [&](int nc, int buf) {
        const int kr = threadIdx.x >> 2;          // 0..31
        const int cb = (threadIdx.x & 3) * 32;    // 0,32,64,96 (halves)
        const _Float16* gk = k + (size_t)(nc + kr) * C2 + cb;
        const unsigned  lk = ks_base + (unsigned)(buf * (32 * 136 * 2)
                                                 + kr * (136 * 2) + cb * 2);
        #pragma unroll
        for (int i = 0; i < 4; ++i)
            async_copy_b128(lk + 16 * i, gk + 8 * i);

        const int c = threadIdx.x;                // 0..127
        const _Float16* gv = vT + (size_t)c * NN + nc;
        const unsigned  lv = vts_base + (unsigned)(buf * (C2 * 32 * 2) + c * 64);
        #pragma unroll
        for (int i = 0; i < 4; ++i)
            async_copy_b128(lv + 16 * i, gv + 8 * i);
    };

    // resident Q strip: 4 A-fragments (K = 0..127 in steps of 32)
    v16h qa[4];
    {
        const _Float16* qrow = q + (size_t)(nw + l15) * C2;
        #pragma unroll
        for (int f = 0; f < 4; ++f) {
            const int c0 = 32 * f;
            #pragma unroll
            for (int j = 0; j < 4; ++j) {
                int kk = c0 + 16 * g + 2 * j;
                qa[f][2 * j]     = qrow[kk];
                qa[f][2 * j + 1] = qrow[kk + 1];
                int kh = c0 + 16 * g + 8 + 2 * j;
                qa[f][8 + 2 * j]     = qrow[kh];
                qa[f][8 + 2 * j + 1] = qrow[kh + 1];
            }
        }
    }

    v8f   oacc[8] = {};
    float mrow[8], lrow[8];
    #pragma unroll
    for (int r = 0; r < 8; ++r) { mrow[r] = -3.0e38f; lrow[r] = 0.0f; }

    issue_chunk(0, 0);   // prologue prefetch

    for (int nc = 0; nc < NN; nc += 32) {
        const int buf = (nc >> 5) & 1;
        wait_async0();        // this wave's chunk-nc data is in LDS
        __syncthreads();      // everyone's data in; everyone done with buf^1
        if (nc + 32 < NN) issue_chunk(nc + 32, buf ^ 1);

        // S tiles: s0 = keys nc..nc+15, s1 = keys nc+16..nc+31
        v8f s0 = {}, s1 = {};
        #pragma unroll
        for (int f = 0; f < 4; ++f) {
            const int c0 = 32 * f;
            const _Float16* kp0 = &ks[buf][l15][c0 + 16 * g];
            const _Float16* kp1 = &ks[buf][16 + l15][c0 + 16 * g];
            v16h kb0, kb1;
            #pragma unroll
            for (int j = 0; j < 16; ++j) { kb0[j] = kp0[j]; kb1[j] = kp1[j]; }
            s0 = __builtin_amdgcn_wmma_f32_16x16x32_f16(
                false, qa[f], false, kb0, (short)0, s0, false, false);
            s1 = __builtin_amdgcn_wmma_f32_16x16x32_f16(
                false, qa[f], false, kb1, (short)0, s1, false, false);
        }

        // online softmax: row = r + 8g, cols spread over 16-lane group
        float rescale[8];
        #pragma unroll
        for (int r = 0; r < 8; ++r) {
            float mx = fmaxf(s0[r], s1[r]);
            #pragma unroll
            for (int off = 1; off < 16; off <<= 1)
                mx = fmaxf(mx, __shfl_xor(mx, off, 32));
            const float mnew = fmaxf(mrow[r], mx);
            const float sc   = __expf(mrow[r] - mnew);
            mrow[r] = mnew;
            const float p0 = __expf(s0[r] - mnew);
            const float p1 = __expf(s1[r] - mnew);
            s0[r] = p0; s1[r] = p1;
            float rs = p0 + p1;
            #pragma unroll
            for (int off = 1; off < 16; off <<= 1)
                rs += __shfl_xor(rs, off, 32);
            lrow[r] = lrow[r] * sc + rs;
            rescale[r] = sc;
        }
        #pragma unroll
        for (int t = 0; t < 8; ++t)
            #pragma unroll
            for (int r = 0; r < 8; ++r)
                oacc[t][r] *= rescale[r];

        // P (D layout) -> A layout via per-wave LDS bounce.
        // Wave-local: LDS ops from one wave complete in order; only a
        // scheduling fence is needed, not a block barrier.
        #pragma unroll
        for (int r = 0; r < 8; ++r) {
            ps[wave][r + 8 * g][l15]      = (_Float16)s0[r];
            ps[wave][r + 8 * g][16 + l15] = (_Float16)s1[r];
        }
        __builtin_amdgcn_wave_barrier();

        v16h pa;
        #pragma unroll
        for (int j = 0; j < 4; ++j) {
            int kk = 16 * g + 2 * j;
            pa[2 * j]     = ps[wave][l15][kk];
            pa[2 * j + 1] = ps[wave][l15][kk + 1];
            int kh = 16 * g + 8 + 2 * j;
            pa[8 + 2 * j]     = ps[wave][l15][kh];
            pa[8 + 2 * j + 1] = ps[wave][l15][kh + 1];
        }

        // O += P @ V   (8 c-tiles, K = 32 keys)
        #pragma unroll
        for (int ct = 0; ct < 8; ++ct) {
            const _Float16* vp = &vts[buf][ct * 16 + l15][16 * g];
            v16h vb;
            #pragma unroll
            for (int j = 0; j < 16; ++j) vb[j] = vp[j];
            oacc[ct] = __builtin_amdgcn_wmma_f32_16x16x32_f16(
                false, pa, false, vb, (short)0, oacc[ct], false, false);
        }
    }

    // normalize and bounce O -> A layout (per-wave buffer, wave-local fence)
    #pragma unroll
    for (int t = 0; t < 8; ++t) {
        #pragma unroll
        for (int r = 0; r < 8; ++r)
            os[wave][r + 8 * g][t * 16 + l15] =
                (_Float16)(oacc[t][r] * (1.0f / lrow[r]));
    }
    __builtin_amdgcn_wave_barrier();

    v16h oa[4];
    #pragma unroll
    for (int f = 0; f < 4; ++f) {
        const int c0 = 32 * f;
        #pragma unroll
        for (int j = 0; j < 4; ++j) {
            int kk = c0 + 16 * g + 2 * j;
            oa[f][2 * j]     = os[wave][l15][kk];
            oa[f][2 * j + 1] = os[wave][l15][kk + 1];
            int kh = c0 + 16 * g + 8 + 2 * j;
            oa[f][8 + 2 * j]     = os[wave][l15][kh];
            oa[f][8 + 2 * j + 1] = os[wave][l15][kh + 1];
        }
    }

    // fused output projection: y[b,co,n] = x + b_out[co] + sum_c2 O[n,c2]*w_out[co,c2]
    #pragma unroll
    for (int ct = 0; ct < 16; ++ct) {
        v8f yacc = {};
        #pragma unroll
        for (int f = 0; f < 4; ++f) {
            const _Float16* wp = wo16 + (size_t)(ct * 16 + l15) * C2 + 32 * f + 16 * g;
            v16h wb;
            #pragma unroll
            for (int j = 0; j < 16; ++j) wb[j] = wp[j];
            yacc = __builtin_amdgcn_wmma_f32_16x16x32_f16(
                false, oa[f], false, wb, (short)0, yacc, false, false);
        }
        const int co = ct * 16 + l15;
        const float bo = b_out[co];
        const float* xp = x + ((size_t)b * C + co) * NN;
        float*       yp = y + ((size_t)b * C + co) * NN;
        #pragma unroll
        for (int r = 0; r < 8; ++r) {
            const int n = nw + r + 8 * g;
            yp[n] = xp[n] + bo + yacc[r];
        }
    }
}

// ---------------------------------------------------------------------------
extern "C" void kernel_launch(void* const* d_in, const int* in_sizes, int n_in,
                              void* d_out, int out_size, void* d_ws, size_t ws_size,
                              hipStream_t stream) {
    const float* x       = (const float*)d_in[0];
    const float* w_theta = (const float*)d_in[1];
    const float* b_theta = (const float*)d_in[2];
    const float* w_phi   = (const float*)d_in[3];
    const float* b_phi   = (const float*)d_in[4];
    const float* w_g     = (const float*)d_in[5];
    const float* b_g     = (const float*)d_in[6];
    const float* w_out   = (const float*)d_in[7];
    const float* b_out   = (const float*)d_in[8];
    float* y = (float*)d_out;

    // workspace: [0, 3*B*N*C2) halves = q,k (row-major) + v (transposed);
    // then 131072 halves of f16 weights
    _Float16* qkv = (_Float16*)d_ws;
    _Float16* w16 = qkv + (size_t)3 * BB * NN * C2;

    cvt_weights_kernel<<<131072 / 256, 256, 0, stream>>>(w_theta, w_phi, w_g, w_out, w16);

    proj_kernel<<<dim3(NN / 64, 3, BB), 128, 0, stream>>>(
        x, w16, b_theta, b_phi, b_g, qkv);

    attn_kernel<<<dim3(NN / 64, BB), 128, 0, stream>>>(
        x, qkv, w16 + (size_t)3 * C2 * C, b_out, y);
}
